// OneTransBlock_30794915512696
// MI455X (gfx1250) — matrix operand order repaired and checked by reading, not compiled, not run on hardware
//
#include <hip/hip_runtime.h>
#include <hip/hip_bf16.h>
#include <math.h>

// ---------------------------------------------------------------------------
// Types / WMMA helpers (CDNA5 wave32, v_wmma_f32_16x16x32_bf16)
// ---------------------------------------------------------------------------
typedef __bf16 bf16;
typedef __attribute__((ext_vector_type(16))) __bf16 v16bf;
typedef __attribute__((ext_vector_type(8)))  __bf16 v8bf;
typedef __attribute__((ext_vector_type(4)))  __bf16 v4bf;
typedef __attribute__((ext_vector_type(8)))  float  v8f;
typedef __attribute__((ext_vector_type(4)))  int    v4i_t;

union FragBF { v16bf v; v8bf h[2]; };

#define WMMA_BF16(Afrag, Bfrag, Cacc) \
  __builtin_amdgcn_wmma_f32_16x16x32_bf16(false, (Afrag), false, (Bfrag), (short)0, (Cacc), false, false)

// Async global->LDS DMA (ASYNCcnt path), guarded so absence of the builtins
// falls back to the proven manual load/store path.
#if defined(__has_builtin)
#if __has_builtin(__builtin_amdgcn_global_load_async_to_lds_b128) && \
    __has_builtin(__builtin_amdgcn_s_wait_asynccnt)
#define USE_ASYNC_LDS 1
#endif
#endif
#ifndef USE_ASYNC_LDS
#define USE_ASYNC_LDS 0
#endif

#if USE_ASYNC_LDS
__device__ __forceinline__ void async_copy16(const void* g, void* l) {
  // Builtin expects v4i pointers: AS1 for global src, AS3 for LDS dst.
  // AS1 pointer value == flat global address; AS3 offset == low 32 bits of the
  // generic LDS pointer (ISA 10.2 aperture mapping), so integer casts are exact.
  __attribute__((address_space(1))) v4i_t* gp =
      (__attribute__((address_space(1))) v4i_t*)(unsigned long long)(const char*)g;
  __attribute__((address_space(3))) v4i_t* lp =
      (__attribute__((address_space(3))) v4i_t*)(unsigned int)(unsigned long long)(char*)l;
  __builtin_amdgcn_global_load_async_to_lds_b128(gp, lp, 0, 0);
}
#endif

constexpr int Dm   = 1024;   // d_model
constexpr int Sq   = 2048;   // seq len
constexpr int BATCH= 4;
constexpr int NH   = 16;
constexpr int DEP  = 64;
constexpr int U1c  = 256;
constexpr int NSc  = 8;
constexpr int Ttok = BATCH * Sq;          // 8192 rows
constexpr int TAIL0= Sq - NSc;            // 2040

__device__ __forceinline__ float silu_f(float x) {
  return x / (1.0f + __expf(-x));
}

// ---------------------------------------------------------------------------
// RMSNorm: row of 1024 f32 -> f32 + bf16 copies
// ---------------------------------------------------------------------------
__global__ void __launch_bounds__(256)
k_rmsnorm(const float* __restrict__ x, const float* __restrict__ w,
          float* __restrict__ y32, bf16* __restrict__ y16)
{
  const int row = blockIdx.x, tid = threadIdx.x;
  const float4 v = ((const float4*)(x + (size_t)row * Dm))[tid];
  float ss = v.x*v.x + v.y*v.y + v.z*v.z + v.w*v.w;
#pragma unroll
  for (int m = 16; m >= 1; m >>= 1) ss += __shfl_xor(ss, m, 32);
  __shared__ float red[8];
  if ((tid & 31) == 0) red[tid >> 5] = ss;
  __syncthreads();
  const float tot = red[0]+red[1]+red[2]+red[3]+red[4]+red[5]+red[6]+red[7];
  const float rms = rsqrtf(tot * (1.0f / Dm) + 1e-6f);
  const float4 wv = ((const float4*)w)[tid];
  float4 o;
  o.x = v.x*rms*wv.x; o.y = v.y*rms*wv.y; o.z = v.z*rms*wv.z; o.w = v.w*rms*wv.w;
  ((float4*)(y32 + (size_t)row * Dm))[tid] = o;
  v4bf ob; ob[0]=(bf16)o.x; ob[1]=(bf16)o.y; ob[2]=(bf16)o.z; ob[3]=(bf16)o.w;
  ((v4bf*)(y16 + (size_t)row * Dm))[tid] = ob;
}

// ---------------------------------------------------------------------------
// Tiled WMMA GEMM, double-buffered LDS, async A-tile staging.
// C[M x NTOT] = A_bf16[M x KDIM] * B_f32[KDIM x NTOT] (+epilogue)
// Block tile 128x64, 256 threads = 8 waves, each wave a 32x32 sub-tile (2x2 WMMA).
// B converted f32->bf16 while staging to LDS (transposed: Bl[n][k]).
// MODE 0: QKV -> bf16 [p][b][h][s][d] (V transposed [d][s]); bias = bkqv grp8
// MODE 1: x1 = acc + bd + h32                              -> f32
// MODE 2: f1 = silu(acc + b1)                              -> bf16
// MODE 3: out = x1 + silu(acc + b2)                        -> f32
// ---------------------------------------------------------------------------
template<int MODE, int KDIM, int NTOT>
__global__ void __launch_bounds__(256)
k_gemm_wmma(const bf16*  __restrict__ A,
            const float* __restrict__ Bsrc, int ldb,
            const float* __restrict__ bias,
            const float* __restrict__ res,
            void* __restrict__ outp)
{
  __shared__ bf16 Al[2][128][40];   // padded to break bank conflicts
  __shared__ bf16 Bl[2][64][40];    // transposed: Bl[buf][n][k]
  const int tid  = threadIdx.x;
  const int lane = tid & 31;
  const int hlf  = (lane >> 4) & 1;
  const int ln   = lane & 15;
  const int wave = tid >> 5;
  const int m0   = blockIdx.x * 128;
  const int n0   = blockIdx.y * 64;

  const float* Bt;
  if constexpr (MODE == 0) {
    const int p = n0 >> 10, col = n0 & 1023;
    Bt = Bsrc + (size_t)p * Dm * Dm + col;
  } else {
    Bt = Bsrc + n0;
  }

  const int mt = (wave & 3) * 2;   // m-tile base (of 8)
  const int nt = (wave >> 2) * 2;  // n-tile base (of 4)

  // loader thread mappings
  const int rA = tid >> 1, cA = (tid & 1) * 16;   // A: 128 rows x 32 halves
  const int kk = tid >> 3, nn = (tid & 7) * 8;    // B: 32 k-rows x 64 cols

  auto load_tiles = [&](int k0, int buf) {
    const bf16* ga = A + (size_t)(m0 + rA) * KDIM + k0 + cA;
#if USE_ASYNC_LDS
    async_copy16(ga,     &Al[buf][rA][cA]);
    async_copy16(ga + 8, &Al[buf][rA][cA + 8]);
#else
    const v8bf* gav = (const v8bf*)ga;
    v8bf a0 = gav[0], a1 = gav[1];
    *(v8bf*)&Al[buf][rA][cA]     = a0;
    *(v8bf*)&Al[buf][rA][cA + 8] = a1;
#endif
    const float4* g = (const float4*)(Bt + (size_t)(k0 + kk) * ldb + nn);
    const float4 f0 = g[0], f1 = g[1];
    Bl[buf][nn + 0][kk] = (bf16)f0.x; Bl[buf][nn + 1][kk] = (bf16)f0.y;
    Bl[buf][nn + 2][kk] = (bf16)f0.z; Bl[buf][nn + 3][kk] = (bf16)f0.w;
    Bl[buf][nn + 4][kk] = (bf16)f1.x; Bl[buf][nn + 5][kk] = (bf16)f1.y;
    Bl[buf][nn + 6][kk] = (bf16)f1.z; Bl[buf][nn + 7][kk] = (bf16)f1.w;
    if (k0 + 32 < KDIM)
      __builtin_prefetch((const void*)(Bt + (size_t)(k0 + 32 + kk) * ldb + nn), 0, 0);
  };

  v8f acc[2][2] = {};

  load_tiles(0, 0);
  for (int k0 = 0; k0 < KDIM; k0 += 32) {
    const int buf = (k0 >> 5) & 1;
#if USE_ASYNC_LDS
    __builtin_amdgcn_s_wait_asynccnt(0);   // my async stores to LDS landed
#endif
    __syncthreads();                       // everyone's tile[buf] is ready
    if (k0 + 32 < KDIM) load_tiles(k0 + 32, buf ^ 1);  // prefetch next tile

    FragBF a0, a1, b0, b1;
    a0.h[0] = *(const v8bf*)&Al[buf][(mt+0)*16 + ln][hlf*8];
    a0.h[1] = *(const v8bf*)&Al[buf][(mt+0)*16 + ln][hlf*8 + 16];
    a1.h[0] = *(const v8bf*)&Al[buf][(mt+1)*16 + ln][hlf*8];
    a1.h[1] = *(const v8bf*)&Al[buf][(mt+1)*16 + ln][hlf*8 + 16];
    b0.h[0] = *(const v8bf*)&Bl[buf][(nt+0)*16 + ln][hlf*8];
    b0.h[1] = *(const v8bf*)&Bl[buf][(nt+0)*16 + ln][hlf*8 + 16];
    b1.h[0] = *(const v8bf*)&Bl[buf][(nt+1)*16 + ln][hlf*8];
    b1.h[1] = *(const v8bf*)&Bl[buf][(nt+1)*16 + ln][hlf*8 + 16];

    acc[0][0] = WMMA_BF16(a0.v, b0.v, acc[0][0]);
    acc[0][1] = WMMA_BF16(a0.v, b1.v, acc[0][1]);
    acc[1][0] = WMMA_BF16(a1.v, b0.v, acc[1][0]);
    acc[1][1] = WMMA_BF16(a1.v, b1.v, acc[1][1]);
  }

  // Epilogue
#pragma unroll
  for (int im = 0; im < 2; ++im) {
#pragma unroll
    for (int in_ = 0; in_ < 2; ++in_) {
      const int gn = n0 + (nt + in_) * 16 + ln;
      const float bn = bias[gn];
      v8f c = acc[im][in_];
#pragma unroll
      for (int r = 0; r < 8; ++r) {
        const int m = m0 + (mt + im) * 16 + r + 8 * hlf;
        float val = c[r] + bn;
        if constexpr (MODE == 0) {
          const int p = gn >> 10, cc = gn & 1023, hh = cc >> 6, dd = cc & 63;
          const int b = m >> 11, s = m & 2047;
          bf16* q = (bf16*)outp;
          size_t off;
          if (p == 2) off = ((size_t)(2*BATCH + b)*NH + hh)*((size_t)Sq*DEP) + (size_t)dd*Sq + s;
          else        off = ((size_t)(p*BATCH + b)*NH + hh)*((size_t)Sq*DEP) + (size_t)s*DEP + dd;
          q[off] = (bf16)val;
        } else if constexpr (MODE == 1) {
          val += res[(size_t)m * Dm + gn];
          ((float*)outp)[(size_t)m * Dm + gn] = val;
        } else if constexpr (MODE == 2) {
          ((bf16*)outp)[(size_t)m * U1c + gn] = (bf16)silu_f(val);
        } else {
          ((float*)outp)[(size_t)m * Dm + gn] =
              res[(size_t)m * Dm + gn] + silu_f(val);
        }
      }
    }
  }
}

// ---------------------------------------------------------------------------
// Flash attention: one wave per 16-row Q tile, 32-key steps, bf16 WMMA,
// online softmax with shfl_xor row reductions. K rows [s][d], V transposed [d][s].
// ---------------------------------------------------------------------------
__global__ void __launch_bounds__(128)
k_flash_attn(const bf16* __restrict__ qkv, bf16* __restrict__ attn)
{
  const int tid  = threadIdx.x;
  const int lane = tid & 31;
  const int hlf  = (lane >> 4) & 1;
  const int ln   = lane & 15;
  const int wave = tid >> 5;
  const int hd   = blockIdx.y, b = blockIdx.z;
  const int q0   = (blockIdx.x * 4 + wave) * 16;

  const bf16* kb = qkv + ((size_t)(0*BATCH + b)*NH + hd) * ((size_t)Sq * DEP);
  const bf16* qb = qkv + ((size_t)(1*BATCH + b)*NH + hd) * ((size_t)Sq * DEP);
  const bf16* vt = qkv + ((size_t)(2*BATCH + b)*NH + hd) * ((size_t)Sq * DEP); // [64][S]

  __shared__ bf16 plds[4][16][40];
  bf16 (*P)[40] = plds[wave];

  FragBF qa[2];
#pragma unroll
  for (int sl = 0; sl < 2; ++sl) {
    const bf16* src = qb + (size_t)(q0 + ln) * DEP + sl * 32 + hlf * 8;
    qa[sl].h[0] = *(const v8bf*)src;
    qa[sl].h[1] = *(const v8bf*)(src + 16);
  }

  v8f o[4] = {};
  float mrow[8], lrow[8];
#pragma unroll
  for (int r = 0; r < 8; ++r) { mrow[r] = -1e30f; lrow[r] = 0.0f; }

  const int nk = (q0 + 15) / 32 + 1;
  for (int kt = 0; kt < nk; ++kt) {
    const int s0 = kt * 32;
    v8f sc[2] = {};
#pragma unroll
    for (int t = 0; t < 2; ++t) {
#pragma unroll
      for (int sl = 0; sl < 2; ++sl) {
        FragBF kf;
        const bf16* src = kb + (size_t)(s0 + t*16 + ln) * DEP + sl * 32 + hlf * 8;
        kf.h[0] = *(const v8bf*)src;
        kf.h[1] = *(const v8bf*)(src + 16);
        sc[t] = WMMA_BF16(qa[sl].v, kf.v, sc[t]);
      }
    }
    // scale + causal mask
    const bool msk = (s0 + 31) > q0;
#pragma unroll
    for (int t = 0; t < 2; ++t) {
#pragma unroll
      for (int r = 0; r < 8; ++r) {
        float v = sc[t][r] * 0.125f;  // 1/sqrt(64)
        if (msk) {
          const int key = s0 + t*16 + ln;
          const int qq  = q0 + r + 8*hlf;
          if (key > qq) v = -1e30f;
        }
        sc[t][r] = v;
      }
    }
    // online softmax (row = r + 8*hlf, 16-lane reductions stay within half)
#pragma unroll
    for (int r = 0; r < 8; ++r) {
      float mx = fmaxf(sc[0][r], sc[1][r]);
#pragma unroll
      for (int mkk = 1; mkk <= 8; mkk <<= 1) mx = fmaxf(mx, __shfl_xor(mx, mkk, 32));
      const float mnew  = fmaxf(mrow[r], mx);
      const float alpha = __expf(mrow[r] - mnew);
      mrow[r] = mnew;
      const float p0 = __expf(sc[0][r] - mnew);
      const float p1 = __expf(sc[1][r] - mnew);
      float rs = p0 + p1;
#pragma unroll
      for (int mkk = 1; mkk <= 8; mkk <<= 1) rs += __shfl_xor(rs, mkk, 32);
      lrow[r] = lrow[r] * alpha + rs;
#pragma unroll
      for (int t = 0; t < 4; ++t) o[t][r] *= alpha;
      P[r + 8*hlf][ln]      = (bf16)p0;
      P[r + 8*hlf][16 + ln] = (bf16)p1;
    }
    __builtin_amdgcn_wave_barrier();
    FragBF pf;
    pf.h[0] = *(const v8bf*)&P[ln][hlf*8];
    pf.h[1] = *(const v8bf*)&P[ln][hlf*8 + 16];
#pragma unroll
    for (int t = 0; t < 4; ++t) {
      FragBF vf;
      const bf16* src = vt + (size_t)(t*16 + ln) * Sq + s0 + hlf * 8;
      vf.h[0] = *(const v8bf*)src;
      vf.h[1] = *(const v8bf*)(src + 16);
      o[t] = WMMA_BF16(pf.v, vf.v, o[t]);
    }
    __builtin_amdgcn_wave_barrier();
  }

#pragma unroll
  for (int r = 0; r < 8; ++r) {
    const float inv = 1.0f / lrow[r];
    const size_t row = (size_t)(b * Sq + q0 + r + 8*hlf);
#pragma unroll
    for (int t = 0; t < 4; ++t)
      attn[row * Dm + hd * DEP + t*16 + ln] = (bf16)(o[t][r] * inv);
  }
}

// ---------------------------------------------------------------------------
// Tail fix-ups (grouped per-token weights for the last NS tokens of each batch)
// ---------------------------------------------------------------------------
__global__ void __launch_bounds__(256)
k_tail_qkv(const float* __restrict__ h32, const float* __restrict__ Wkqv,
           const float* __restrict__ bkqv, bf16* __restrict__ qkv)
{
  const int bx = blockIdx.x;                     // b*24 + i*3 + p
  const int b = bx / (NSc * 3), rem = bx % (NSc * 3);
  const int i = rem / 3, p = rem % 3;
  const int c = blockIdx.y * 256 + threadIdx.x;
  const int s = TAIL0 + i;
  const size_t tok = (size_t)b * Sq + s;
  __shared__ float hr[Dm];
#pragma unroll
  for (int j = 0; j < 4; ++j) hr[threadIdx.x + j*256] = h32[tok*Dm + threadIdx.x + j*256];
  __syncthreads();
  float acc = bkqv[(size_t)(i*3 + p) * Dm + c];
  const float* wcol = Wkqv + (size_t)(i*3 + p) * Dm * Dm + c;
  for (int k = 0; k < Dm; ++k) acc += hr[k] * wcol[(size_t)k * Dm];
  const int hh = c >> 6, dd = c & 63;
  size_t off;
  if (p == 2) off = ((size_t)(2*BATCH + b)*NH + hh)*((size_t)Sq*DEP) + (size_t)dd*Sq + s;
  else        off = ((size_t)(p*BATCH + b)*NH + hh)*((size_t)Sq*DEP) + (size_t)s*DEP + dd;
  qkv[off] = (bf16)acc;
}

__global__ void __launch_bounds__(256)
k_tail_ffn1(const float* __restrict__ h2_32, const float* __restrict__ W1,
            const float* __restrict__ b1, bf16* __restrict__ f1,
            float* __restrict__ tailf1)
{
  const int bx = blockIdx.x;                     // b*8 + i
  const int b = bx >> 3, i = bx & 7;
  const int c = threadIdx.x;                     // 0..255
  const int s = TAIL0 + i;
  const size_t tok = (size_t)b * Sq + s;
  __shared__ float hr[Dm];
#pragma unroll
  for (int j = 0; j < 4; ++j) hr[c + j*256] = h2_32[tok*Dm + c + j*256];
  __syncthreads();
  float acc = b1[(size_t)i * U1c + c];
  const float* wcol = W1 + (size_t)i * Dm * U1c + c;
  for (int k = 0; k < Dm; ++k) acc += hr[k] * wcol[(size_t)k * U1c];
  const float y = silu_f(acc);
  f1[tok * U1c + c] = (bf16)y;
  tailf1[(size_t)bx * U1c + c] = y;
}

__global__ void __launch_bounds__(256)
k_tail_ffn2(const float* __restrict__ tailf1, const float* __restrict__ W2,
            const float* __restrict__ b2, const float* __restrict__ x1,
            float* __restrict__ out)
{
  const int bx = blockIdx.x;                     // b*8 + i
  const int b = bx >> 3, i = bx & 7;
  const int c = blockIdx.y * 256 + threadIdx.x;
  const int s = TAIL0 + i;
  const size_t tok = (size_t)b * Sq + s;
  __shared__ float fr[U1c];
  fr[threadIdx.x] = tailf1[(size_t)bx * U1c + threadIdx.x];
  __syncthreads();
  float acc = b2[(size_t)i * Dm + c];
  const float* wcol = W2 + (size_t)i * U1c * Dm + c;
  for (int k = 0; k < U1c; ++k) acc += fr[k] * wcol[(size_t)k * Dm];
  out[tok * Dm + c] = x1[tok * Dm + c] + silu_f(acc);
}

// ---------------------------------------------------------------------------
// Host orchestration
// ---------------------------------------------------------------------------
extern "C" void kernel_launch(void* const* d_in, const int* in_sizes, int n_in,
                              void* d_out, int out_size, void* d_ws, size_t ws_size,
                              hipStream_t stream) {
  (void)in_sizes; (void)n_in; (void)out_size; (void)ws_size;
  const float* x     = (const float*)d_in[0];
  const float* Wkqv  = (const float*)d_in[1];
  const float* bkqv  = (const float*)d_in[2];
  const float* Wd    = (const float*)d_in[3];
  const float* bd    = (const float*)d_in[4];
  const float* rms0w = (const float*)d_in[5];
  const float* rms1w = (const float*)d_in[6];
  const float* W1    = (const float*)d_in[7];
  const float* b1    = (const float*)d_in[8];
  const float* W2    = (const float*)d_in[9];
  const float* b2    = (const float*)d_in[10];
  float* out = (float*)d_out;

  // Workspace layout (128MB + 32KB), with lifetime-based aliasing:
  //   [0,32M)    h_f32            -> later h2_f32
  //   [32M,48M)  h_bf16           -> later attn_bf16
  //   [48M,96M)  qkv_bf16 (48MB)  -> later h2_bf16 (16MB) @48M, f1_bf16 (4MB) @64M
  //   [96M,128M) x1_f32
  //   [128M,+32K) tail f1 scratch
  char* ws = (char*)d_ws;
  constexpr size_t MB = 1ull << 20;
  float* h32      = (float*)(ws);
  bf16*  h16      = (bf16*) (ws + 32*MB);
  bf16*  qkv16    = (bf16*) (ws + 48*MB);
  float* x1       = (float*)(ws + 96*MB);
  float* tailf1   = (float*)(ws + 128*MB);
  bf16*  attn16   = h16;                  // alias (h16 dead after QKV GEMM)
  float* h2_32    = h32;                  // alias (h32 dead after out-proj)
  bf16*  h2_16    = qkv16;                // alias (qkv dead after attention)
  bf16*  f1_16    = (bf16*)(ws + 64*MB);  // alias inside old qkv region

  // 1) h = rmsnorm(x)
  k_rmsnorm<<<dim3(Ttok), dim3(256), 0, stream>>>(x, rms0w, h32, h16);

  // 2) QKV with shared group-8 weights (bulk of tokens), bf16 WMMA
  k_gemm_wmma<0, Dm, 3*Dm><<<dim3(Ttok/128, 3*Dm/64), dim3(256), 0, stream>>>(
      h16, Wkqv + (size_t)NSc*3*Dm*Dm, Dm, bkqv + (size_t)NSc*3*Dm, nullptr, qkv16);

  // 3) Tail tokens: per-token grouped weights (overwrite 8 rows/batch/proj)
  k_tail_qkv<<<dim3(BATCH*NSc*3, Dm/256), dim3(256), 0, stream>>>(h32, Wkqv, bkqv, qkv16);

  // 4) Causal flash attention (bf16 WMMA, online softmax)
  k_flash_attn<<<dim3(Sq/64, NH, BATCH), dim3(128), 0, stream>>>(qkv16, attn16);

  // 5) x1 = h + attn @ Wd + bd
  k_gemm_wmma<1, Dm, Dm><<<dim3(Ttok/128, Dm/64), dim3(256), 0, stream>>>(
      attn16, Wd, Dm, bd, h32, x1);

  // 6) h2 = rmsnorm(x1)
  k_rmsnorm<<<dim3(Ttok), dim3(256), 0, stream>>>(x1, rms1w, h2_32, h2_16);

  // 7) f1 = silu(h2 @ W1[8] + b1[8])
  k_gemm_wmma<2, Dm, U1c><<<dim3(Ttok/128, U1c/64), dim3(256), 0, stream>>>(
      h2_16, W1 + (size_t)NSc*Dm*U1c, U1c, b1 + (size_t)NSc*U1c, nullptr, f1_16);

  // 8) Tail FFN1 (grouped weights) — must finish before FFN2 consumes f1
  k_tail_ffn1<<<dim3(BATCH*NSc), dim3(256), 0, stream>>>(h2_32, W1, b1, f1_16, tailf1);

  // 9) out = x1 + silu(f1 @ W2[8] + b2[8])
  k_gemm_wmma<3, U1c, Dm><<<dim3(Ttok/128, Dm/64), dim3(256), 0, stream>>>(
      f1_16, W2 + (size_t)NSc*U1c*Dm, Dm, b2 + (size_t)NSc*Dm, x1, out);

  // 10) Tail FFN2 (grouped weights) overwrites the tail rows of the output
  k_tail_ffn2<<<dim3(BATCH*NSc, Dm/256), dim3(256), 0, stream>>>(tailf1, W2, b2, x1, out);
}
